// LocalityEnhancedSelfAttention_73710228734875
// MI455X (gfx1250) — matrix operand order repaired and checked by reading, not compiled, not run on hardware
//
#include <hip/hip_runtime.h>
#include <hip/hip_bf16.h>

typedef _Float16 v16h __attribute__((ext_vector_type(16)));
typedef float    v8f  __attribute__((ext_vector_type(8)));
typedef _Float16 h4   __attribute__((ext_vector_type(4)));
typedef _Float16 h8   __attribute__((ext_vector_type(8)));
typedef float    f4v  __attribute__((ext_vector_type(4)));

#define BATCH 8
#define NTOK  1025          // 32*32 + 1 (cls)
#define JPAD  1056          // NTOK padded to multiple of 32
#define CDIM  512
#define INNER 512
#define HEADS 8
#define DH    64
#define FH    32
#define FW    17
#define SPAT  1024
#define ROWS  (BATCH * NTOK)        // 8200
#define SROWS (BATCH * SPAT)        // 8192
#define HWCN  (SPAT * CDIM)         // 524288
#define LSTR  40                    // LDS stride in halves (20 DW, bank-clean)
#define MTB   256                   // GEMM block M tile (8 waves x 32 rows)

__device__ __forceinline__ float gelu_exact(float x) {
    return 0.5f * x * (1.0f + erff(x * 0.70710678118654752f));
}

__device__ __forceinline__ v16h cat_h8(h8 lo, h8 hi_) {
    v16h a;
#pragma unroll
    for (int i = 0; i < 8; ++i) { a[i] = lo[i]; a[i + 8] = hi_[i]; }
    return a;
}

// ---- weight transpose + f16 convert: Bt[n*K + k] = B[k*N + n] -------------
__global__ void wtrans(const float* __restrict__ B, _Float16* __restrict__ Bt,
                       int K, int N) {
    int idx = blockIdx.x * blockDim.x + threadIdx.x;
    if (idx >= K * N) return;
    int k = idx / N, n = idx - k * N;
    Bt[(size_t)n * K + k] = (_Float16)B[idx];
}

// ---- tiled GEMM: C(M,N) = A(M,K) @ B(K,N), A fp32, B pre-transposed f16 ---
// block = 256 threads (8 waves); block tile 256(M) x 64(N); wave tile 32x64.
// double-buffered LDS pipeline: next tile's global loads issue before compute.
// mode: 0 = f32 out (+bias/gelu), 1 = f16 out, 2 = f16 scatter into Vt[b,h,d,j]
__global__ __launch_bounds__(256) void gemm_wmma_v3(
    const float* __restrict__ A, int lda,
    const _Float16* __restrict__ Bt,
    void* __restrict__ Cout, int ldc,
    int M, int N, int K,
    const float* __restrict__ bias, int act, int mode) {
    const int tid = threadIdx.x;
    const int wave = tid >> 5, lane = tid & 31;
    const int r = lane & 15, hi = lane >> 4;
    const int m0b = blockIdx.x * MTB;
    const int n0 = blockIdx.y * 64;

    __shared__ _Float16 As[2][MTB * LSTR];   // 40 KB
    __shared__ _Float16 Bs[2][64 * LSTR];    // 10 KB

    v8f acc0[4], acc1[4];
#pragma unroll
    for (int c = 0; c < 4; ++c)
#pragma unroll
        for (int e = 0; e < 8; ++e) { acc0[c][e] = 0.0f; acc1[c][e] = 0.0f; }

    const int bcol = tid >> 2, bpart = tid & 3;        // B fill mapping
    const int gcol = min(n0 + bcol, N - 1);
    const _Float16* btcol = Bt + (size_t)gcol * K + bpart * 8;

    // staged tile in registers
    f4v areg[8];
    h8  breg;

    auto gload = [&](int k0) {
#pragma unroll
        for (int it = 0; it < 8; ++it) {
            int s = tid + it * 256;                    // 0..2047
            int row = s >> 3, kk = (s & 7) * 4;
            int grow = min(m0b + row, M - 1);
            const float* gp = A + (size_t)grow * lda + k0 + kk;
            __builtin_prefetch(gp + 32, 0, 1);
            areg[it] = *(const f4v*)gp;
        }
        breg = *(const h8*)(btcol + k0);
    };
    auto lstore = [&](int buf) {
#pragma unroll
        for (int it = 0; it < 8; ++it) {
            int s = tid + it * 256;
            int row = s >> 3, kk = (s & 7) * 4;
            h4 hv;
#pragma unroll
            for (int j = 0; j < 4; ++j) hv[j] = (_Float16)areg[it][j];
            *(h4*)(&As[buf][row * LSTR + kk]) = hv;
        }
        *(h8*)(&Bs[buf][bcol * LSTR + bpart * 8]) = breg;
    };

    const int nsteps = K >> 5;
    gload(0);
    lstore(0);
    __syncthreads();

    for (int step = 0; step < nsteps; ++step) {
        const int cur = step & 1;
        if (step + 1 < nsteps) gload((step + 1) << 5);

        const _Float16* ap0 = &As[cur][(wave * 32 + r) * LSTR];
        const _Float16* ap1 = ap0 + 16 * LSTR;
        v16h a0 = cat_h8(*(const h8*)(ap0 + hi * 16), *(const h8*)(ap0 + 8 + hi * 16));
        v16h a1 = cat_h8(*(const h8*)(ap1 + hi * 16), *(const h8*)(ap1 + 8 + hi * 16));
#pragma unroll
        for (int c = 0; c < 4; ++c) {
            const _Float16* bp = &Bs[cur][(c * 16 + r) * LSTR + hi * 16];
            v16h bf = cat_h8(*(const h8*)bp, *(const h8*)(bp + 8));
            acc0[c] = __builtin_amdgcn_wmma_f32_16x16x32_f16(
                false, a0, false, bf, (short)0, acc0[c], false, false);
            acc1[c] = __builtin_amdgcn_wmma_f32_16x16x32_f16(
                false, a1, false, bf, (short)0, acc1[c], false, false);
        }
        if (step + 1 < nsteps) lstore((step + 1) & 1);
        __syncthreads();
    }

    // ---- epilogue ---------------------------------------------------------
#pragma unroll
    for (int half = 0; half < 2; ++half) {
#pragma unroll
        for (int c = 0; c < 4; ++c) {
#pragma unroll
            for (int rr = 0; rr < 8; ++rr) {
                int row = m0b + wave * 32 + half * 16 + rr + hi * 8;
                int col = n0 + c * 16 + r;
                if (row < M && col < N) {
                    float v = half ? acc1[c][rr] : acc0[c][rr];
                    if (bias) v += bias[col];
                    if (act == 1) v = gelu_exact(v);
                    if (mode == 0) {
                        ((float*)Cout)[(size_t)row * ldc + col] = v;
                    } else if (mode == 1) {
                        ((_Float16*)Cout)[(size_t)row * ldc + col] = (_Float16)v;
                    } else {  // mode 2: scatter into Vt[b][h][d][j], f16
                        int b = row / NTOK, n = row - b * NTOK;
                        int h = col >> 6, d = col & 63;
                        ((_Float16*)Cout)[(((size_t)(b * HEADS + h) * DH + d) * JPAD) + n] =
                            (_Float16)v;
                    }
                }
            }
        }
    }
}

// ---- Flash attention: one wave = 16 query rows of one (b,h) ---------------
// qh,kh: (B,N,HEADS,DH) f16; vt: (B,H,DH,JPAD) f16. out -> cat[...,h*64+d]
__global__ __launch_bounds__(32) void attn_flash(
    const _Float16* __restrict__ qh, const _Float16* __restrict__ kh,
    const _Float16* __restrict__ vt, float* __restrict__ cat) {
    const int lane = threadIdx.x & 31;
    const int r = lane & 15, hi = lane >> 4;
    const int mt = blockIdx.x;
    const int bh = blockIdx.y;
    const int b = bh >> 3, h = bh & 7;
    const float scale = 0.125f;

    const int mrow = min(mt * 16 + r, NTOK - 1);
    const _Float16* qrow = qh + ((size_t)(b * NTOK + mrow) * HEADS + h) * DH;
    v16h aq0 = cat_h8(*(const h8*)(qrow + hi * 16), *(const h8*)(qrow + 8 + hi * 16));
    v16h aq1 = cat_h8(*(const h8*)(qrow + 32 + hi * 16), *(const h8*)(qrow + 40 + hi * 16));

    v8f acc[4];
    float mi[8], li[8];
#pragma unroll
    for (int c = 0; c < 4; ++c)
#pragma unroll
        for (int e = 0; e < 8; ++e) acc[c][e] = 0.0f;
#pragma unroll
    for (int rr = 0; rr < 8; ++rr) { mi[rr] = -1e30f; li[rr] = 0.0f; }

    __shared__ _Float16 plds[16 * LSTR];
    const _Float16* vbase = vt + (size_t)(b * HEADS + h) * DH * JPAD;

    for (int j0 = 0; j0 < NTOK; j0 += 32) {
        v8f s[2];
#pragma unroll
        for (int t = 0; t < 2; ++t) {
#pragma unroll
            for (int e = 0; e < 8; ++e) s[t][e] = 0.0f;
            const int jc = min(j0 + t * 16 + r, NTOK - 1);
            const _Float16* krow = kh + ((size_t)(b * NTOK + jc) * HEADS + h) * DH + hi * 16;
            v16h bk0 = cat_h8(*(const h8*)krow, *(const h8*)(krow + 8));
            v16h bk1 = cat_h8(*(const h8*)(krow + 32), *(const h8*)(krow + 40));
            s[t] = __builtin_amdgcn_wmma_f32_16x16x32_f16(false, aq0, false, bk0, (short)0, s[t], false, false);
            s[t] = __builtin_amdgcn_wmma_f32_16x16x32_f16(false, aq1, false, bk1, (short)0, s[t], false, false);
        }
        const bool mask0 = (j0 + r) >= NTOK;
        const bool mask1 = (j0 + 16 + r) >= NTOK;

        float sc0[8], sc1[8], mnew[8], alpha[8];
#pragma unroll
        for (int rr = 0; rr < 8; ++rr) {
            sc0[rr] = mask0 ? -3.0e38f : s[0][rr] * scale;
            sc1[rr] = mask1 ? -3.0e38f : s[1][rr] * scale;
            float vmax = fmaxf(sc0[rr], sc1[rr]);
#pragma unroll
            for (int off = 8; off > 0; off >>= 1)
                vmax = fmaxf(vmax, __shfl_xor(vmax, off, 16));
            mnew[rr] = fmaxf(mi[rr], vmax);
            alpha[rr] = __expf(mi[rr] - mnew[rr]);
            mi[rr] = mnew[rr];
            li[rr] *= alpha[rr];
        }
#pragma unroll
        for (int c = 0; c < 4; ++c)
#pragma unroll
            for (int rr = 0; rr < 8; ++rr) acc[c][rr] *= alpha[rr];

#pragma unroll
        for (int rr = 0; rr < 8; ++rr) {
            float p0 = __expf(sc0[rr] - mnew[rr]);
            float p1 = __expf(sc1[rr] - mnew[rr]);
            float rs = p0 + p1;
#pragma unroll
            for (int off = 8; off > 0; off >>= 1) rs += __shfl_xor(rs, off, 16);
            li[rr] += rs;
            plds[(rr + hi * 8) * LSTR + r]      = (_Float16)p0;
            plds[(rr + hi * 8) * LSTR + 16 + r] = (_Float16)p1;
        }
        __syncthreads();

        const _Float16* prow = &plds[r * LSTR];
        v16h ap = cat_h8(*(const h8*)(prow + hi * 16), *(const h8*)(prow + 8 + hi * 16));
#pragma unroll
        for (int c = 0; c < 4; ++c) {
            const _Float16* vp = vbase + (size_t)(c * 16 + r) * JPAD + j0 + hi * 16;
            v16h bv = cat_h8(*(const h8*)vp, *(const h8*)(vp + 8));
            acc[c] = __builtin_amdgcn_wmma_f32_16x16x32_f16(false, ap, false, bv, (short)0, acc[c], false, false);
        }
        __syncthreads();
    }

#pragma unroll
    for (int c = 0; c < 4; ++c)
#pragma unroll
        for (int rr = 0; rr < 8; ++rr) {
            int row = mt * 16 + rr + hi * 8;
            if (row < NTOK)
                cat[(size_t)(b * NTOK + row) * (3 * INNER) + h * 64 + c * 16 + r] =
                    acc[c][rr] / li[rr];
        }
}

// ---- Depthwise 3x3 SAME conv on (B,32,32,C) -------------------------------
__global__ void dwconv3x3(const float* __restrict__ x, const float* __restrict__ w,
                          float* __restrict__ out) {
    int idx = blockIdx.x * blockDim.x + threadIdx.x;
    if (idx >= BATCH * SPAT * CDIM) return;
    int ch = idx & 511;
    int p = (idx >> 9) & 1023;
    int b = idx >> 19;
    int hh = p >> 5, ww = p & 31;
    float s = 0.0f;
#pragma unroll
    for (int kh = 0; kh < 3; ++kh)
#pragma unroll
        for (int kw = 0; kw < 3; ++kw) {
            int ih = hh + kh - 1, iw = ww + kw - 1;
            if (ih >= 0 && ih < 32 && iw >= 0 && iw < 32)
                s += x[((size_t)(b * NTOK) + 1 + ih * 32 + iw) * CDIM + ch] *
                     w[(kh * 3 + kw) * CDIM + ch];
        }
    out[idx] = s;
}

// ---- Frequency branch: xe = irfft2(rfft2(sp) * (wr + i wi)), ortho --------
__global__ __launch_bounds__(256) void fft_filter(
    const float* __restrict__ x, const float* __restrict__ wr,
    const float* __restrict__ wi, float* __restrict__ out) {
    const int bc = blockIdx.x;
    const int ch = bc & 511, b = bc >> 9;
    const int tid = threadIdx.x;
    __shared__ float img[SPAT];
    __shared__ float Fr[FH * FW], Fi[FH * FW], Gr[FH * FW], Gi[FH * FW];

    for (int p = tid; p < SPAT; p += 256)
        img[p] = x[((size_t)(b * NTOK) + 1 + p) * CDIM + ch];
    __syncthreads();

    const float c2pi = 6.28318530717958647f / 32.0f;

    for (int p = tid; p < FH * FW; p += 256) {
        int h = p / FW, fw = p % FW;
        float sr = 0.0f, si = 0.0f;
        for (int w = 0; w < 32; ++w) {
            float sn, cs;
            __sincosf(c2pi * (float)(w * fw), &sn, &cs);
            float v = img[h * 32 + w];
            sr += v * cs;
            si -= v * sn;
        }
        Fr[p] = sr; Fi[p] = si;
    }
    __syncthreads();

    for (int p = tid; p < FH * FW; p += 256) {
        int fh = p / FW, fw = p % FW;
        float zr = 0.0f, zi = 0.0f;
        for (int h = 0; h < 32; ++h) {
            float sn, cs;
            __sincosf(c2pi * (float)(h * fh), &sn, &cs);
            float fr = Fr[h * FW + fw], fi = Fi[h * FW + fw];
            zr += fr * cs + fi * sn;
            zi += fi * cs - fr * sn;
        }
        float wrv = wr[(size_t)p * CDIM + ch];
        float wiv = wi[(size_t)p * CDIM + ch];
        Gr[p] = zr * wrv - zi * wiv;
        Gi[p] = zr * wiv + zi * wrv;
    }
    __syncthreads();

    for (int p = tid; p < SPAT; p += 256) {
        int h = p >> 5, w = p & 31;
        float s = 0.0f;
        for (int fh = 0; fh < 32; ++fh)
            for (int fw = 0; fw < FW; ++fw) {
                float wgt = (fw == 0 || fw == 16) ? 1.0f : 2.0f;
                float sn, cs;
                __sincosf(c2pi * (float)(h * fh + w * fw), &sn, &cs);
                s += wgt * (Gr[fh * FW + fw] * cs - Gi[fh * FW + fw] * sn);
            }
        out[((size_t)(b * SPAT) + p) * CDIM + ch] = s * (1.0f / 1024.0f);
    }
}

// ---- GroupNorm(1) stats ---------------------------------------------------
__global__ __launch_bounds__(256) void gn_stats(const float* __restrict__ xin,
                                                float* __restrict__ red) {
    const int b = blockIdx.x >> 4, chunk = blockIdx.x & 15;
    const int tid = threadIdx.x;
    const float* base = xin + (size_t)b * HWCN + (size_t)chunk * (HWCN / 16);
    float s = 0.0f, s2 = 0.0f;
    for (int i = tid; i < HWCN / 16; i += 256) {
        float v = base[i];
        s += v; s2 += v * v;
    }
    __shared__ float sh0[256], sh1[256];
    sh0[tid] = s; sh1[tid] = s2;
    __syncthreads();
    for (int st = 128; st > 0; st >>= 1) {
        if (tid < st) { sh0[tid] += sh0[tid + st]; sh1[tid] += sh1[tid + st]; }
        __syncthreads();
    }
    if (tid == 0) {
        atomicAdd(&red[b * 2], sh0[0]);
        atomicAdd(&red[b * 2 + 1], sh1[0]);
    }
}

__global__ void gn_apply(const float* __restrict__ xin, const float* __restrict__ red,
                         const float* __restrict__ g, const float* __restrict__ bb,
                         float* __restrict__ cat, int coff, int act) {
    int idx = blockIdx.x * blockDim.x + threadIdx.x;
    if (idx >= BATCH * SPAT * CDIM) return;
    int ch = idx & 511;
    int s = (idx >> 9) & 1023;
    int b = idx >> 19;
    float mean = red[b * 2] * (1.0f / (float)HWCN);
    float var = red[b * 2 + 1] * (1.0f / (float)HWCN) - mean * mean;
    float y = (xin[idx] - mean) * rsqrtf(var + 1e-5f) * g[ch] + bb[ch];
    if (act == 1) y = gelu_exact(y);
    cat[((size_t)(b * NTOK) + 1 + s) * (3 * INNER) + coff + ch] = y;
}

__global__ void cls_mean(float* __restrict__ cat, int coff) {
    int t = blockIdx.x * blockDim.x + threadIdx.x;
    if (t >= BATCH * CDIM) return;
    int ch = t & 511, b = t >> 9;
    float s = 0.0f;
    for (int p = 0; p < SPAT; ++p)
        s += cat[((size_t)(b * NTOK) + 1 + p) * (3 * INNER) + coff + ch];
    cat[(size_t)(b * NTOK) * (3 * INNER) + coff + ch] = s * (1.0f / (float)SPAT);
}

// ---- Gate layer 2 (384->3) + softmax + blend ------------------------------
__global__ void gate_fuse(const float* __restrict__ g1, const float* __restrict__ w2,
                          const float* __restrict__ b2, const float* __restrict__ cat,
                          float* __restrict__ fused) {
    int row = blockIdx.x * blockDim.x + threadIdx.x;
    if (row >= ROWS) return;
    const float* gr = g1 + (size_t)row * 384;
    float l0 = b2[0], l1 = b2[1], l2 = b2[2];
    for (int i = 0; i < 384; ++i) {
        float v = gr[i];
        l0 += v * w2[i * 3 + 0];
        l1 += v * w2[i * 3 + 1];
        l2 += v * w2[i * 3 + 2];
    }
    float mx = fmaxf(l0, fmaxf(l1, l2));
    float e0 = __expf(l0 - mx), e1 = __expf(l1 - mx), e2 = __expf(l2 - mx);
    float inv = 1.0f / (e0 + e1 + e2);
    e0 *= inv; e1 *= inv; e2 *= inv;
    const float* cr = cat + (size_t)row * (3 * INNER);
    float* fo = fused + (size_t)row * INNER;
    for (int ch = 0; ch < INNER; ++ch)
        fo[ch] = e0 * cr[ch] + e1 * cr[INNER + ch] + e2 * cr[2 * INNER + ch];
}

// ---------------------------------------------------------------------------
extern "C" void kernel_launch(void* const* d_in, const int* in_sizes, int n_in,
                              void* d_out, int out_size, void* d_ws, size_t ws_size,
                              hipStream_t stream) {
    const float* x     = (const float*)d_in[0];
    const float* Wq    = (const float*)d_in[1];
    const float* Wk    = (const float*)d_in[2];
    const float* Wv    = (const float*)d_in[3];
    const float* dw_w  = (const float*)d_in[4];
    const float* pw_w  = (const float*)d_in[5];
    const float* ln_g  = (const float*)d_in[6];
    const float* ln_b  = (const float*)d_in[7];
    const float* fr_wr = (const float*)d_in[8];
    const float* fr_wi = (const float*)d_in[9];
    const float* fr_pw = (const float*)d_in[10];
    const float* fr_g  = (const float*)d_in[11];
    const float* fr_b  = (const float*)d_in[12];
    const float* g_W1  = (const float*)d_in[13];
    const float* g_b1  = (const float*)d_in[14];
    const float* g_W2  = (const float*)d_in[15];
    const float* g_b2  = (const float*)d_in[16];
    const float* Wp    = (const float*)d_in[17];
    const float* bp    = (const float*)d_in[18];
    float* out = (float*)d_out;

    // workspace layout (float units)
    float* ws   = (float*)d_ws;
    float* rq   = ws;                                   // qh (f16) region
    float* rk   = rq   + (size_t)ROWS * INNER;          // kh (f16) region
    float* rv   = rk   + (size_t)ROWS * INNER;          // Vt (f16) region
    float* cat  = rv   + (size_t)ROWS * INNER;          // 8200*1536 f32
    float* tmpA = cat  + (size_t)ROWS * 3 * INNER;      // 8200*512 f32
    float* tmpB = tmpA + (size_t)ROWS * INNER;          // 8192*512 f32
    float* g1o  = tmpB + (size_t)SROWS * INNER;         // 8200*384 f32
    float* red  = g1o  + (size_t)ROWS * 384;            // 16 f32
    _Float16* bt = (_Float16*)(red + 16);               // up to 1536*384 halves

    _Float16* qh = (_Float16*)rq;
    _Float16* kh = (_Float16*)rk;
    _Float16* vt = (_Float16*)rv;

    auto gemm = [&](const float* A, int lda, const float* Bw, void* C, int ldc,
                    int M, int N, int K, const float* bias, int act, int mode) {
        wtrans<<<(K * N + 255) / 256, 256, 0, stream>>>(Bw, bt, K, N);
        dim3 g((M + MTB - 1) / MTB, (N + 63) / 64);
        gemm_wmma_v3<<<g, 256, 0, stream>>>(A, lda, bt, C, ldc, M, N, K, bias, act, mode);
    };

    // Branch 1: QKV (f16 outputs) + flash attention -> cat[:, 0:512]
    gemm(x, CDIM, Wq, qh, INNER, ROWS, INNER, CDIM, nullptr, 0, 1);
    gemm(x, CDIM, Wk, kh, INNER, ROWS, INNER, CDIM, nullptr, 0, 1);
    gemm(x, CDIM, Wv, vt, INNER, ROWS, INNER, CDIM, nullptr, 0, 2);  // scatter Vt
    attn_flash<<<dim3((NTOK + 15) / 16, BATCH * HEADS), 32, 0, stream>>>(qh, kh, vt, cat);

    // Branch 2: dwconv -> pointwise -> GroupNorm -> cat[:, 512:1024]
    dwconv3x3<<<(BATCH * SPAT * CDIM + 255) / 256, 256, 0, stream>>>(x, dw_w, tmpA);
    gemm(tmpA, CDIM, pw_w, tmpB, INNER, SROWS, INNER, CDIM, nullptr, 0, 0);
    hipMemsetAsync(red, 0, 16 * sizeof(float), stream);
    gn_stats<<<BATCH * 16, 256, 0, stream>>>(tmpB, red);
    gn_apply<<<(BATCH * SPAT * CDIM + 255) / 256, 256, 0, stream>>>(tmpB, red, ln_g, ln_b, cat, INNER, 0);
    cls_mean<<<(BATCH * CDIM + 255) / 256, 256, 0, stream>>>(cat, INNER);

    // Branch 3: spectral filter -> pointwise -> GroupNorm+GELU -> cat[:, 1024:1536]
    fft_filter<<<BATCH * CDIM, 256, 0, stream>>>(x, fr_wr, fr_wi, tmpA);
    gemm(tmpA, CDIM, fr_pw, tmpB, INNER, SROWS, INNER, CDIM, nullptr, 0, 0);
    hipMemsetAsync(red, 0, 16 * sizeof(float), stream);
    gn_stats<<<BATCH * 16, 256, 0, stream>>>(tmpB, red);
    gn_apply<<<(BATCH * SPAT * CDIM + 255) / 256, 256, 0, stream>>>(tmpB, red, fr_g, fr_b, cat, 2 * INNER, 1);
    cls_mean<<<(BATCH * CDIM + 255) / 256, 256, 0, stream>>>(cat, 2 * INNER);

    // Gate MLP: cat @ W1 (+b1, gelu) -> g1o ; then W2 + softmax + blend -> tmpA
    gemm(cat, 3 * INNER, g_W1, g1o, 3 * INNER / 4, ROWS, 3 * INNER / 4, 3 * INNER, g_b1, 1, 0);
    gate_fuse<<<(ROWS + 255) / 256, 256, 0, stream>>>(g1o, g_W2, g_b2, cat, tmpA);

    // Output projection
    gemm(tmpA, INNER, Wp, out, CDIM, ROWS, CDIM, INNER, bp, 0, 0);
}
